// WaterManagementGNN_8778913153101
// MI455X (gfx1250) — compile-verified, matrix operand order checked by reading, not run on hardware
//
#include <hip/hip_runtime.h>
#include <hip/hip_bf16.h>

typedef __bf16 bf16_t;
typedef __attribute__((ext_vector_type(16))) __bf16 v16bf;
typedef __attribute__((ext_vector_type(8)))  float  v8f;

#define N_NODES 6144
#define T_STEPS 24
#define H_DIM   128
#define E_EDGES 196608

// ---------------------------------------------------------------------------
// Elementwise helpers
// ---------------------------------------------------------------------------
__global__ void k_zero_f32(float* __restrict__ p, long n) {
  long i = (long)blockIdx.x * 256 + threadIdx.x;
  if (i < n) p[i] = 0.0f;
}

__global__ void k_zero_bf(bf16_t* __restrict__ p, long n) {
  long i = (long)blockIdx.x * 256 + threadIdx.x;
  if (i < n) p[i] = (bf16_t)0.0f;
}

// f32 -> bf16 with optional column zero-padding ([rows, incols] -> [rows, outcols])
__global__ void k_cvt_pad(const float* __restrict__ in, bf16_t* __restrict__ out,
                          int rows, int incols, int outcols) {
  long idx = (long)blockIdx.x * 256 + threadIdx.x;
  long total = (long)rows * outcols;
  if (idx >= total) return;
  int r = (int)(idx / outcols);
  int c = (int)(idx % outcols);
  float v = (c < incols) ? in[(long)r * incols + c] : 0.0f;
  out[idx] = (bf16_t)v;
}

__global__ void k_deg(const int* __restrict__ dst, float* __restrict__ deg, int E) {
  int e = blockIdx.x * 256 + threadIdx.x;
  if (e < E) atomicAdd(&deg[dst[e]], 1.0f);
}

__global__ void k_dis(float* __restrict__ deg, int n) {
  int i = blockIdx.x * 256 + threadIdx.x;
  if (i < n) deg[i] = rsqrtf(deg[i] + 1.0f);
}

// edge scatter: agg[dst] += h[src] * dis[src]*dis[dst]  (4 feats per thread)
__global__ void k_scatter(const float* __restrict__ hpre,
                          const int* __restrict__ src, const int* __restrict__ dst,
                          const float* __restrict__ dis, float* __restrict__ agg, int E) {
  long tid = (long)blockIdx.x * 256 + threadIdx.x;
  long e = tid >> 5;
  if (e >= E) return;
  int f4 = (int)(tid & 31) << 2;
  int s = src[e], d = dst[e];
  float nrm = dis[s] * dis[d];
  const float4 hv = *(const float4*)(hpre + (long)s * H_DIM + f4);
  float* base = agg + (long)d * H_DIM + f4;
  atomicAdd(base + 0, hv.x * nrm);
  atomicAdd(base + 1, hv.y * nrm);
  atomicAdd(base + 2, hv.z * nrm);
  atomicAdd(base + 3, hv.w * nrm);
}

// out = agg + hpre*dis^2 + bias  (+relu), dual f32/bf16 output
__global__ void k_gcn_combine(const float* __restrict__ agg, const float* __restrict__ hpre,
                              const float* __restrict__ dis, const float* __restrict__ b,
                              float* __restrict__ outf, bf16_t* __restrict__ outbf, int relu) {
  long idx = (long)blockIdx.x * 256 + threadIdx.x;
  if (idx >= (long)N_NODES * H_DIM) return;
  int n = (int)(idx >> 7), f = (int)(idx & 127);
  float dd = dis[n];
  float v = agg[idx] + hpre[idx] * dd * dd + b[f];
  if (relu) v = fmaxf(v, 0.0f);
  outf[idx] = v;
  if (outbf) outbf[idx] = (bf16_t)v;
}

// LSTM gate math: gates[N,512] (i,f,g,o) -> c,h  (h in f32 + bf16)
__global__ void k_lstm_gate(const float* __restrict__ gates,
                            const float* __restrict__ bih, const float* __restrict__ bhh,
                            float* __restrict__ c, float* __restrict__ hf,
                            bf16_t* __restrict__ hb) {
  long idx = (long)blockIdx.x * 256 + threadIdx.x;
  if (idx >= (long)N_NODES * H_DIM) return;
  int n = (int)(idx >> 7), j = (int)(idx & 127);
  const float* g = gates + (long)n * 512;
  float gi = g[j]       + bih[j]       + bhh[j];
  float gf = g[128 + j] + bih[128 + j] + bhh[128 + j];
  float gg = g[256 + j] + bih[256 + j] + bhh[256 + j];
  float go = g[384 + j] + bih[384 + j] + bhh[384 + j];
  float si = 1.0f / (1.0f + __expf(-gi));
  float sf = 1.0f / (1.0f + __expf(-gf));
  float so = 1.0f / (1.0f + __expf(-go));
  float cn = sf * c[idx] + si * tanhf(gg);
  c[idx] = cn;
  float hn = so * tanhf(cn);
  hf[idx] = hn;
  hb[idx] = (bf16_t)hn;
}

// h = a + b, dual output
__global__ void k_add_cvt(const float* __restrict__ a, const float* __restrict__ b,
                          float* __restrict__ outf, bf16_t* __restrict__ outbf, long n) {
  long i = (long)blockIdx.x * 256 + threadIdx.x;
  if (i >= n) return;
  float v = a[i] + b[i];
  outf[i] = v;
  outbf[i] = (bf16_t)v;
}

// ---------------------------------------------------------------------------
// WMMA GEMM: Cout[M,Ncol] = act( A[M,K](bf16,lda) @ W[Ncol,K]^T + Cin + bias )
// Block: 256 thr = 8 waves (4x2), tile 128x64; wave tile 32x32 = four 16x16
// D-tiles (2 A-frags x 2 B-frags, each fragment reused twice). Double-buffered
// LDS (one barrier per K-step); register prefetch of the next K-tile overlaps
// global loads with WMMA issue.
// ---------------------------------------------------------------------------
__global__ __launch_bounds__(256) void k_gemm(
    const bf16_t* __restrict__ A, int lda,
    const bf16_t* __restrict__ W,
    const float* __restrict__ Cin,
    const float* __restrict__ bias,
    float* __restrict__ Cout,
    bf16_t* __restrict__ Cbf,
    int M, int Ncol, int K, int act) {
  __shared__ bf16_t As[2][128][32];
  __shared__ bf16_t Bs[2][64][32];
  const int tid  = threadIdx.x;
  const int lane = tid & 31;
  const int wave = tid >> 5;
  const int wm = wave >> 1;           // 0..3  (M sub-block, 32 rows)
  const int wn = wave & 1;            // 0..1  (N sub-block, 32 cols)
  const int bm0 = blockIdx.x * 128;
  const int bn0 = blockIdx.y * 64;

  v8f acc00 = {}, acc01 = {}, acc10 = {}, acc11 = {};

  // staging: A tile 128x32 halves -> 2 x uint4 per thread; B tile 64x32 -> 1 x uint4
  const int alr = tid >> 1;             // A stage row 0..127
  const int als = (tid & 1) * 16;       // A stage k-seg {0,16}
  const int blr = tid >> 2;             // B stage row 0..63
  const int bls = (tid & 3) * 8;        // B stage k-seg {0,8,16,24}
  const int ar = bm0 + alr;
  const int br = bn0 + blr;

  const int khalf = (lane >> 4) * 8;
  const int arow = wm * 32 + (lane & 15);
  const int bcol = wn * 32 + (lane & 15);

  uint4 ra0, ra1, rb0;
  auto stage_load = [&](int k0) {
    ra0 = make_uint4(0u, 0u, 0u, 0u);
    ra1 = make_uint4(0u, 0u, 0u, 0u);
    rb0 = make_uint4(0u, 0u, 0u, 0u);
    if (ar < M) {
      const bf16_t* ap = A + (long)ar * lda + k0 + als;
      ra0 = *(const uint4*)(ap);
      ra1 = *(const uint4*)(ap + 8);
    }
    if (br < Ncol) rb0 = *(const uint4*)(W + (long)br * K + k0 + bls);
  };
  auto stage_store = [&](int buf) {
    *(uint4*)(&As[buf][alr][als])     = ra0;
    *(uint4*)(&As[buf][alr][als + 8]) = ra1;
    *(uint4*)(&Bs[buf][blr][bls])     = rb0;
  };

  const int nk = K >> 5;
  stage_load(0);
  stage_store(0);

  for (int ki = 0; ki < nk; ++ki) {
    __syncthreads();
    const int cur = ki & 1;
    const bool more = (ki + 1) < nk;
    if (more) {
      stage_load((ki + 1) << 5);
      if (ki + 2 < nk) {  // gfx1250 global_prefetch of the K-tile after next
        if (ar < M)    __builtin_prefetch(A + (long)ar * lda + ((ki + 2) << 5) + als, 0, 1);
        if (br < Ncol) __builtin_prefetch(W + (long)br * K   + ((ki + 2) << 5) + bls, 0, 1);
      }
    }

    v16bf af0, af1, bf0, bf1;
#pragma unroll
    for (int r = 0; r < 8; ++r) {
      int kb = (r >> 2) * 16 + (r & 3) * 2 + khalf;
      af0[2 * r]     = As[cur][arow][kb];
      af0[2 * r + 1] = As[cur][arow][kb + 1];
      af1[2 * r]     = As[cur][arow + 16][kb];
      af1[2 * r + 1] = As[cur][arow + 16][kb + 1];
      bf0[2 * r]     = Bs[cur][bcol][kb];
      bf0[2 * r + 1] = Bs[cur][bcol][kb + 1];
      bf1[2 * r]     = Bs[cur][bcol + 16][kb];
      bf1[2 * r + 1] = Bs[cur][bcol + 16][kb + 1];
    }
    acc00 = __builtin_amdgcn_wmma_f32_16x16x32_bf16(false, af0, false, bf0, (short)0, acc00, false, false);
    acc01 = __builtin_amdgcn_wmma_f32_16x16x32_bf16(false, af0, false, bf1, (short)0, acc01, false, false);
    acc10 = __builtin_amdgcn_wmma_f32_16x16x32_bf16(false, af1, false, bf0, (short)0, acc10, false, false);
    acc11 = __builtin_amdgcn_wmma_f32_16x16x32_bf16(false, af1, false, bf1, (short)0, acc11, false, false);

    if (more) stage_store(cur ^ 1);
  }

  // epilogue: D-tile layout row = v + 8*(lane/16), col = lane%16
  const int g = lane >> 4;
  const int cbase = bn0 + wn * 32 + (lane & 15);
  v8f* accs[2][2] = {{&acc00, &acc01}, {&acc10, &acc11}};
#pragma unroll
  for (int sn = 0; sn < 2; ++sn) {
    int cn = cbase + sn * 16;
    if (cn >= Ncol) continue;
    float bb = bias ? bias[cn] : 0.0f;
#pragma unroll
    for (int sm = 0; sm < 2; ++sm) {
      v8f& acc = *accs[sm][sn];
#pragma unroll
      for (int v = 0; v < 8; ++v) {
        int rm = bm0 + wm * 32 + sm * 16 + v + 8 * g;
        if (rm < M) {
          long idx = (long)rm * Ncol + cn;
          float val = acc[v] + bb + (Cin ? Cin[idx] : 0.0f);
          if (act == 1) val = fmaxf(val, 0.0f);
          else if (act == 2) val = 1.0f / (1.0f + __expf(-val));
          Cout[idx] = val;
          if (Cbf) Cbf[idx] = (bf16_t)val;
        }
      }
    }
  }
}

// ---------------------------------------------------------------------------
// Flash attention: 4 heads x d=32, seq n. One wave per (qblock,head).
// qkv: bf16 [n,384] (q|k|v). obf: bf16 [n,128] = softmax(QK^T/sqrt(32)) V.
// Scores and PV both via v_wmma_f32_16x16x32_bf16; P relayout D->A via LDS.
// ---------------------------------------------------------------------------
__global__ __launch_bounds__(128) void k_attn(const bf16_t* __restrict__ qkv,
                                              bf16_t* __restrict__ obf, int n) {
  __shared__ bf16_t Ps[4][16][32];
  const int lane  = threadIdx.x & 31;
  const int hd    = threadIdx.x >> 5;
  const int q0    = blockIdx.x * 16;
  const int m     = lane & 15;
  const int g     = lane >> 4;
  const int khalf = g * 8;
  const float sc  = 0.17677669529663687f;  // 1/sqrt(32)

  v16bf qf;
  {
    const bf16_t* qb = qkv + (long)(q0 + m) * 384 + hd * 32;
#pragma unroll
    for (int r = 0; r < 8; ++r) {
      int kb = (r >> 2) * 16 + (r & 3) * 2 + khalf;
      qf[2 * r] = qb[kb]; qf[2 * r + 1] = qb[kb + 1];
    }
  }

  v8f o0 = {}; v8f o1 = {};
  float mr[8], sr[8];
#pragma unroll
  for (int v = 0; v < 8; ++v) { mr[v] = -1e30f; sr[v] = 0.0f; }

  for (int kb0 = 0; kb0 < n; kb0 += 32) {
    v8f s0 = {}; v8f s1 = {};
    {
      v16bf kf;
      const bf16_t* kp = qkv + (long)(kb0 + m) * 384 + 128 + hd * 32;
#pragma unroll
      for (int r = 0; r < 8; ++r) {
        int kk = (r >> 2) * 16 + (r & 3) * 2 + khalf;
        kf[2 * r] = kp[kk]; kf[2 * r + 1] = kp[kk + 1];
      }
      s0 = __builtin_amdgcn_wmma_f32_16x16x32_bf16(false, qf, false, kf, (short)0, s0, false, false);
    }
    {
      v16bf kf;
      const bf16_t* kp = qkv + (long)(kb0 + 16 + m) * 384 + 128 + hd * 32;
#pragma unroll
      for (int r = 0; r < 8; ++r) {
        int kk = (r >> 2) * 16 + (r & 3) * 2 + khalf;
        kf[2 * r] = kp[kk]; kf[2 * r + 1] = kp[kk + 1];
      }
      s1 = __builtin_amdgcn_wmma_f32_16x16x32_bf16(false, qf, false, kf, (short)0, s1, false, false);
    }

    float pa[8], pb[8];
#pragma unroll
    for (int v = 0; v < 8; ++v) {
      float a = s0[v] * sc, b = s1[v] * sc;
      float tm = fmaxf(a, b);
      tm = fmaxf(tm, __shfl_xor(tm, 1, 32));
      tm = fmaxf(tm, __shfl_xor(tm, 2, 32));
      tm = fmaxf(tm, __shfl_xor(tm, 4, 32));
      tm = fmaxf(tm, __shfl_xor(tm, 8, 32));
      float mn = fmaxf(mr[v], tm);
      float scale = __expf(mr[v] - mn);
      float ea = __expf(a - mn), eb = __expf(b - mn);
      float rs = ea + eb;
      rs += __shfl_xor(rs, 1, 32);
      rs += __shfl_xor(rs, 2, 32);
      rs += __shfl_xor(rs, 4, 32);
      rs += __shfl_xor(rs, 8, 32);
      sr[v] = sr[v] * scale + rs;
      mr[v] = mn;
      o0[v] *= scale; o1[v] *= scale;
      pa[v] = ea; pb[v] = eb;
    }
#pragma unroll
    for (int v = 0; v < 8; ++v) {
      Ps[hd][v + 8 * g][m]      = (bf16_t)pa[v];
      Ps[hd][v + 8 * g][16 + m] = (bf16_t)pb[v];
    }
    __syncthreads();

    v16bf pf, vf0, vf1;
#pragma unroll
    for (int r = 0; r < 8; ++r) {
      int kk = (r >> 2) * 16 + (r & 3) * 2 + khalf;
      pf[2 * r] = Ps[hd][m][kk]; pf[2 * r + 1] = Ps[hd][m][kk + 1];
      const bf16_t* v0p = qkv + (long)(kb0 + kk) * 384 + 256 + hd * 32 + m;
      const bf16_t* v1p = v0p + 384;
      vf0[2 * r] = v0p[0];  vf0[2 * r + 1] = v1p[0];
      vf1[2 * r] = v0p[16]; vf1[2 * r + 1] = v1p[16];
    }
    o0 = __builtin_amdgcn_wmma_f32_16x16x32_bf16(false, pf, false, vf0, (short)0, o0, false, false);
    o1 = __builtin_amdgcn_wmma_f32_16x16x32_bf16(false, pf, false, vf1, (short)0, o1, false, false);
    __syncthreads();
  }

#pragma unroll
  for (int v = 0; v < 8; ++v) {
    float inv = 1.0f / sr[v];
    int row = q0 + v + 8 * g;
    obf[(long)row * 128 + hd * 32 + m]      = (bf16_t)(o0[v] * inv);
    obf[(long)row * 128 + hd * 32 + 16 + m] = (bf16_t)(o1[v] * inv);
  }
}

// ---------------------------------------------------------------------------
// Host orchestration
// ---------------------------------------------------------------------------
extern "C" void kernel_launch(void* const* d_in, const int* in_sizes, int n_in,
                              void* d_out, int out_size, void* d_ws, size_t ws_size,
                              hipStream_t stream) {
  (void)in_sizes; (void)n_in; (void)out_size; (void)ws_size;
  const int N = N_NODES, T = T_STEPS, H = H_DIM, E = E_EDGES;

  const float* x    = (const float*)d_in[0];
  const int*   ei   = (const int*)d_in[1];
  const float* td   = (const float*)d_in[2];
  const float* W1   = (const float*)d_in[3];
  const float* b1   = (const float*)d_in[4];
  const float* W2   = (const float*)d_in[5];
  const float* b2   = (const float*)d_in[6];
  const float* W3   = (const float*)d_in[7];
  const float* b3   = (const float*)d_in[8];
  const float* Wih0 = (const float*)d_in[9];
  const float* Whh0 = (const float*)d_in[10];
  const float* bih0 = (const float*)d_in[11];
  const float* bhh0 = (const float*)d_in[12];
  const float* Wih1 = (const float*)d_in[13];
  const float* Whh1 = (const float*)d_in[14];
  const float* bih1 = (const float*)d_in[15];
  const float* bhh1 = (const float*)d_in[16];
  const float* Win  = (const float*)d_in[17];
  const float* b_in = (const float*)d_in[18];
  const float* Wout = (const float*)d_in[19];
  const float* bout = (const float*)d_in[20];
  const float* Wd1  = (const float*)d_in[21];
  const float* bd1  = (const float*)d_in[22];
  const float* Wd2  = (const float*)d_in[23];
  const float* bd2  = (const float*)d_in[24];
  const float* Wi1  = (const float*)d_in[25];
  const float* bi1  = (const float*)d_in[26];
  const float* Wi2  = (const float*)d_in[27];
  const float* bi2  = (const float*)d_in[28];

  const int* src = ei;
  const int* dst = ei + E;
  float* out = (float*)d_out;
  float* out_wd = out;                 // [N,7]
  float* out_ie = out + (long)N * 7;   // [N,1]
  float* out_h  = out + (long)N * 8;   // [N,128]

  // workspace layout
  char* ws = (char*)d_ws;
  size_t off = 0;
  auto alloc = [&](size_t bytes) -> char* {
    char* p = ws + off;
    off = (off + bytes + 255) & ~(size_t)255;
    return p;
  };
  float*  dis   = (float*)alloc((size_t)N * 4);
  bf16_t* tempo = (bf16_t*)alloc((size_t)N * T * H * 2);
  bf16_t* xpad  = (bf16_t*)alloc((size_t)N * 64 * 2);
  bf16_t* W1b   = (bf16_t*)alloc(128 * 64 * 2);
  bf16_t* W2b   = (bf16_t*)alloc(128 * 128 * 2);
  bf16_t* W3b   = (bf16_t*)alloc(128 * 128 * 2);
  bf16_t* Wih0b = (bf16_t*)alloc(512 * 128 * 2);
  bf16_t* Whh0b = (bf16_t*)alloc(512 * 128 * 2);
  bf16_t* Wih1b = (bf16_t*)alloc(512 * 128 * 2);
  bf16_t* Whh1b = (bf16_t*)alloc(512 * 128 * 2);
  bf16_t* Winb  = (bf16_t*)alloc(384 * 128 * 2);
  bf16_t* Woutb = (bf16_t*)alloc(128 * 128 * 2);
  bf16_t* Wd1b  = (bf16_t*)alloc(64 * 128 * 2);
  bf16_t* Wd2b  = (bf16_t*)alloc(7 * 64 * 2);
  bf16_t* Wi1b  = (bf16_t*)alloc(64 * 128 * 2);
  bf16_t* Wi2b  = (bf16_t*)alloc(64 * 2);
  float*  F1 = (float*)alloc((size_t)N * H * 4);  // scratch / hpre
  float*  F2 = (float*)alloc((size_t)N * H * 4);  // agg / h-combined
  float*  F3 = (float*)alloc((size_t)N * H * 4);  // gcn out
  float*  F4 = (float*)alloc((size_t)N * H * 4);  // c0
  float*  F5 = (float*)alloc((size_t)N * H * 4);  // c1
  float*  F6 = (float*)alloc((size_t)N * H * 4);  // h1 f32
  float*  G  = (float*)alloc((size_t)N * 512 * 4);
  bf16_t* B1 = (bf16_t*)alloc((size_t)N * H * 2);
  bf16_t* B2 = (bf16_t*)alloc((size_t)N * H * 2);
  bf16_t* B3 = (bf16_t*)alloc((size_t)N * H * 2);
  float*  Q32 = (float*)alloc((size_t)N * 384 * 4);
  bf16_t* QBF = (bf16_t*)alloc((size_t)N * 384 * 2);

  auto blks = [](long n) { return (unsigned)((n + 255) / 256); };
  auto gemm = [&](const bf16_t* A, int lda, const bf16_t* W, const float* Cin,
                  const float* bias, float* Cout, bf16_t* Cbf,
                  int M, int Ncol, int K, int act) {
    dim3 grid((M + 127) / 128, (Ncol + 63) / 64);
    k_gemm<<<grid, 256, 0, stream>>>(A, lda, W, Cin, bias, Cout, Cbf, M, Ncol, K, act);
  };

  // --- conversions ---
  k_cvt_pad<<<blks((long)N * 64), 256, 0, stream>>>(x, xpad, N, 48, 64);
  k_cvt_pad<<<blks(128 * 64), 256, 0, stream>>>(W1, W1b, 128, 48, 64);
  k_cvt_pad<<<blks(128 * 128), 256, 0, stream>>>(W2, W2b, 128 * 128, 1, 1);
  k_cvt_pad<<<blks(128 * 128), 256, 0, stream>>>(W3, W3b, 128 * 128, 1, 1);
  k_cvt_pad<<<blks(512 * 128), 256, 0, stream>>>(Wih0, Wih0b, 512 * 128, 1, 1);
  k_cvt_pad<<<blks(512 * 128), 256, 0, stream>>>(Whh0, Whh0b, 512 * 128, 1, 1);
  k_cvt_pad<<<blks(512 * 128), 256, 0, stream>>>(Wih1, Wih1b, 512 * 128, 1, 1);
  k_cvt_pad<<<blks(512 * 128), 256, 0, stream>>>(Whh1, Whh1b, 512 * 128, 1, 1);
  k_cvt_pad<<<blks(384 * 128), 256, 0, stream>>>(Win, Winb, 384 * 128, 1, 1);
  k_cvt_pad<<<blks(128 * 128), 256, 0, stream>>>(Wout, Woutb, 128 * 128, 1, 1);
  k_cvt_pad<<<blks(64 * 128), 256, 0, stream>>>(Wd1, Wd1b, 64 * 128, 1, 1);
  k_cvt_pad<<<blks(7 * 64), 256, 0, stream>>>(Wd2, Wd2b, 7 * 64, 1, 1);
  k_cvt_pad<<<blks(64 * 128), 256, 0, stream>>>(Wi1, Wi1b, 64 * 128, 1, 1);
  k_cvt_pad<<<blks(64), 256, 0, stream>>>(Wi2, Wi2b, 64, 1, 1);
  k_cvt_pad<<<blks((long)N * T * H), 256, 0, stream>>>(td, tempo, (int)((long)N * T), H, H);

  // --- degree / norm ---
  k_zero_f32<<<blks(N), 256, 0, stream>>>(dis, N);
  k_deg<<<blks(E), 256, 0, stream>>>(dst, dis, E);
  k_dis<<<blks(N), 256, 0, stream>>>(dis, N);

  // --- GCN layers ---
  auto gcn = [&](const bf16_t* Ain, int Kin, const bf16_t* Wb, const float* bias,
                 int relu, bf16_t* outbf) {
    gemm(Ain, Kin, Wb, nullptr, nullptr, F1, nullptr, N, H, Kin, 0);
    k_zero_f32<<<blks((long)N * H), 256, 0, stream>>>(F2, (long)N * H);
    k_scatter<<<blks((long)E * 32), 256, 0, stream>>>(F1, src, dst, dis, F2, E);
    k_gcn_combine<<<blks((long)N * H), 256, 0, stream>>>(F2, F1, dis, bias, F3, outbf, relu);
  };
  gcn(xpad, 64, W1b, b1, 1, B1);
  gcn(B1, 128, W2b, b2, 1, B1);
  gcn(B1, 128, W3b, b3, 0, nullptr);   // F3 = graph features (f32)

  // --- 2-layer LSTM over T steps ---
  k_zero_f32<<<blks((long)N * H), 256, 0, stream>>>(F4, (long)N * H);
  k_zero_f32<<<blks((long)N * H), 256, 0, stream>>>(F5, (long)N * H);
  k_zero_bf<<<blks((long)N * H), 256, 0, stream>>>(B2, (long)N * H);
  k_zero_bf<<<blks((long)N * H), 256, 0, stream>>>(B3, (long)N * H);
  for (int t = 0; t < T; ++t) {
    gemm(tempo + (long)t * H, T * H, Wih0b, nullptr, nullptr, G, nullptr, N, 512, 128, 0);
    gemm(B2, 128, Whh0b, G, nullptr, G, nullptr, N, 512, 128, 0);
    k_lstm_gate<<<blks((long)N * H), 256, 0, stream>>>(G, bih0, bhh0, F4, F1, B2);
    gemm(B2, 128, Wih1b, nullptr, nullptr, G, nullptr, N, 512, 128, 0);
    gemm(B3, 128, Whh1b, G, nullptr, G, nullptr, N, 512, 128, 0);
    k_lstm_gate<<<blks((long)N * H), 256, 0, stream>>>(G, bih1, bhh1, F5, F6, B3);
  }
  // h = gcn_out + last hidden  -> F2 (f32) + B1 (bf16)
  k_add_cvt<<<blks((long)N * H), 256, 0, stream>>>(F3, F6, F2, B1, (long)N * H);

  // --- attention ---
  gemm(B1, 128, Winb, nullptr, b_in, Q32, QBF, N, 384, 128, 0);
  k_attn<<<N / 16, 128, 0, stream>>>(QBF, B2, N);
  // h_final = h + o @ Wout^T + b_out  -> out_h (f32) + B1 (bf16)
  gemm(B2, 128, Woutb, F2, bout, out_h, B1, N, 128, 128, 0);

  // --- heads ---
  gemm(B1, 128, Wd1b, nullptr, bd1, F1, B2, N, 64, 128, 1);
  gemm(B2, 64, Wd2b, nullptr, bd2, out_wd, nullptr, N, 7, 64, 0);
  gemm(B1, 128, Wi1b, nullptr, bi1, F1, B2, N, 64, 128, 1);
  gemm(B2, 64, Wi2b, nullptr, bi2, out_ie, nullptr, N, 1, 64, 2);
}
